// VectorQuantizer_13941463843146
// MI455X (gfx1250) — compile-verified, matrix operand order checked by reading, not run on hardware
//
#include <hip/hip_runtime.h>

typedef __attribute__((ext_vector_type(16))) __bf16          v16bf;
typedef __attribute__((ext_vector_type(8)))  float           v8f;
typedef __attribute__((ext_vector_type(8)))  unsigned short  ushort8;
typedef int gv4i __attribute__((vector_size(16)));   // matches builtin pointee: int __vector(4)

union BFrag { ushort8 h[2]; v16bf v; };

#define N_E      8192
#define E_DIM    256
#define M_TOTAL  32768            // 2*16384
#define NTILES   (N_E / 16)       // 512
#define MB       64               // rows of z per block in the argmin kernel
#define LDST     264              // padded LDS row stride in ushorts (528B -> conflict-free)

// ---- d_out layout (floats, concatenated in return order) ----
#define ZQ_OFF     0ULL
#define LOSS_OFF   8388608ULL                 // M_TOTAL*E_DIM
#define IDX_OFF    8388609ULL
#define AGE_OFF    8421377ULL                 // IDX_OFF + 32768
#define USAGE_OFF  8429569ULL                 // AGE_OFF + 8192

// ---- workspace layout (bytes) ----
#define WS_EMBB    0ULL                       // 8192*256 bf16 = 4 MiB
#define WS_ENORM   4194304ULL                 // 8192 f32
#define WS_HIST    4227072ULL                 // 8192 f32
#define WS_USED    4259840ULL                 // 8192 f32
#define WS_LOSS    4292608ULL                 // 1 f32

#if __has_builtin(__builtin_amdgcn_global_load_async_to_lds_b128)
#define HAVE_ASYNC 1
#else
#define HAVE_ASYNC 0
#endif

__device__ __forceinline__ void wait_async0() {
#if __has_builtin(__builtin_amdgcn_s_wait_asynccnt)
    __builtin_amdgcn_s_wait_asynccnt(0);
#else
    asm volatile("s_wait_asynccnt 0" ::: "memory");
#endif
}

__device__ __forceinline__ unsigned short f2bf(float f) {
    unsigned int u = __float_as_uint(f);
    u += 0x7FFFu + ((u >> 16) & 1u);          // round-to-nearest-even
    return (unsigned short)(u >> 16);
}

// ---------------------------------------------------------------------------
// Kernel 1: emb fp32 -> bf16, enorm = ||e||^2 (fp32); zero hist/used/loss.
// One wave per code row. grid = 1024 blocks x 256 threads.
// ---------------------------------------------------------------------------
__global__ __launch_bounds__(256) void vq_prep(const float* __restrict__ emb,
                                               unsigned short* __restrict__ embb,
                                               float* __restrict__ enorm,
                                               float* __restrict__ hist,
                                               float* __restrict__ used,
                                               float* __restrict__ lossacc) {
    const int lane = threadIdx.x & 31;
    const int w    = threadIdx.x >> 5;
    const int n    = blockIdx.x * 8 + w;

    const float4* ep = (const float4*)(emb + (size_t)n * E_DIM);
    float4 f0 = ep[lane * 2 + 0];
    float4 f1 = ep[lane * 2 + 1];

    ushort8 o = { f2bf(f0.x), f2bf(f0.y), f2bf(f0.z), f2bf(f0.w),
                  f2bf(f1.x), f2bf(f1.y), f2bf(f1.z), f2bf(f1.w) };
    *(ushort8*)(embb + (size_t)n * E_DIM + lane * 8) = o;

    float acc = f0.x*f0.x + f0.y*f0.y + f0.z*f0.z + f0.w*f0.w
              + f1.x*f1.x + f1.y*f1.y + f1.z*f1.z + f1.w*f1.w;
#pragma unroll
    for (int off = 16; off >= 1; off >>= 1)
        acc += __shfl_xor(acc, off, 32);

    if (lane == 0) {
        enorm[n] = acc;
        hist[n]  = 0.0f;
        used[n]  = 0.0f;
    }
    if (blockIdx.x == 0 && threadIdx.x == 0) *lossacc = 0.0f;
}

// ---------------------------------------------------------------------------
// Kernel 2: fused GEMM + argmin over codes.
// 128 threads = 4 waves; block owns MB=64 rows; wave owns 16 rows.
// B tiles double-buffered via GLOBAL_LOAD_ASYNC_TO_LDS (ASYNCcnt) when the
// toolchain declares the builtin; otherwise vector-load staging.
// ---------------------------------------------------------------------------
__global__ __launch_bounds__(128) void vq_argmin(const float* __restrict__ z,
                                                 const unsigned short* __restrict__ embb,
                                                 const float* __restrict__ enorm,
                                                 int* __restrict__ idxo) {
    __shared__ __attribute__((aligned(16))) unsigned short zt[MB * LDST];     // 33.8 KB
#if HAVE_ASYNC
    __shared__ __attribute__((aligned(16))) unsigned short bt[2][16 * LDST];  // 16.9 KB
#else
    __shared__ __attribute__((aligned(16))) unsigned short bt[1][16 * LDST];  //  8.4 KB
#endif

    const int tid  = threadIdx.x;
    const int lane = tid & 31;
    const int w    = tid >> 5;
    const int rowBase = blockIdx.x * MB;

#if HAVE_ASYNC
    // Per-tile staging: 512 x 16B chunks, 4 per thread, straight into LDS.
    auto stage_async = [&](int tile, int buf) {
        const char* gsrc = (const char*)(embb + (size_t)tile * 16 * E_DIM);
#pragma unroll
        for (int j = 0; j < 4; ++j) {
            int i = tid + j * 128;
            int r = i >> 5;                 // 32 chunks per code row
            int c = (i & 31) * 8;
            __builtin_amdgcn_global_load_async_to_lds_b128(
                (gv4i*)(gsrc + (size_t)i * 16),
                (gv4i*)&bt[buf][r * LDST + c],
                0, 0);
        }
    };
    stage_async(0, 0);                      // prologue: tile 0 in flight
#endif

    // ---- stage 64x256 z rows into LDS as bf16 (coalesced float4 reads) ----
    const float4* zg = (const float4*)(z + (size_t)rowBase * E_DIM);
#pragma unroll 1
    for (int i = tid; i < MB * (E_DIM / 4); i += 128) {
        float4 f = zg[i];
        int r = i >> 6;                     // 64 float4 per row
        int c = (i & 63) * 4;
        zt[r * LDST + c + 0] = f2bf(f.x);
        zt[r * LDST + c + 1] = f2bf(f.y);
        zt[r * LDST + c + 2] = f2bf(f.z);
        zt[r * LDST + c + 3] = f2bf(f.w);
    }
    __syncthreads();

    // ---- build A fragments once (wave's 16 rows x K=256 -> 8 v16bf) ----
    const int m = lane & 15;                // M % 16 (A) / N (B, C)
    const int h = lane >> 4;                // lane half selects K sub-groups
    v16bf a[8];
#pragma unroll
    for (int t = 0; t < 8; ++t) {
        BFrag f;
        const unsigned short* base = &zt[(w * 16 + m) * LDST + t * 32 + h * 8];
        f.h[0] = *(const ushort8*)(base);        // K = t*32 + h*8 + [0..7]
        f.h[1] = *(const ushort8*)(base + 16);   // K = t*32 + 16 + h*8 + [0..7]
        a[t] = f.v;
    }

    float minv[8];
    int   mini[8];
#pragma unroll
    for (int r = 0; r < 8; ++r) { minv[r] = 3.4e38f; mini[r] = 0; }

    // ---- sweep all 8192 codes in tiles of 16 ----
    for (int nt = 0; nt < NTILES; ++nt) {
#if HAVE_ASYNC
        wait_async0();                      // our async chunks for tile nt landed
        __syncthreads();                    // everyone's landed; prev compute done
        if (nt + 1 < NTILES)                // overlap next tile with this compute
            stage_async(nt + 1, (nt + 1) & 1);
        const unsigned short* cur = &bt[nt & 1][0];
#else
        __syncthreads();                    // previous tile's reads done
        const ushort8* eg = (const ushort8*)(embb + (size_t)nt * 16 * E_DIM);
#pragma unroll 1
        for (int i = tid; i < 16 * (E_DIM / 8); i += 128) {
            ushort8 v = eg[i];
            int r = i >> 5;
            int c = (i & 31) * 8;
            *(ushort8*)&bt[0][r * LDST + c] = v;
        }
        if (nt + 1 < NTILES)                // global_prefetch_b8 of next tile
            __builtin_prefetch(embb + (size_t)(nt + 1) * 16 * E_DIM + tid * 32, 0, 1);
        __syncthreads();
        const unsigned short* cur = &bt[0][0];
#endif
        // Preload ALL B fragments, then 8 back-to-back WMMAs (no per-step
        // full dscnt stalls between ds_load and v_wmma).
        BFrag bf[8];
#pragma unroll
        for (int t = 0; t < 8; ++t) {
            const unsigned short* base = &cur[m * LDST + t * 32 + h * 8];
            bf[t].h[0] = *(const ushort8*)(base);
            bf[t].h[1] = *(const ushort8*)(base + 16);
        }
        v8f acc = {0.f, 0.f, 0.f, 0.f, 0.f, 0.f, 0.f, 0.f};
#pragma unroll
        for (int t = 0; t < 8; ++t)
            acc = __builtin_amdgcn_wmma_f32_16x16x32_bf16(
                      false, a[t], false, bf[t].v, (short)0, acc, false, false);

        const int col = nt * 16 + m;
        const float en = enorm[col];
#pragma unroll
        for (int r = 0; r < 8; ++r) {
            float s = fmaf(-2.0f, acc[r], en);   // ||z||^2 constant per row: drop it
            if (s < minv[r]) { minv[r] = s; mini[r] = col; }
        }
    }

    // ---- cross-lane argmin over the 16-lane N dimension (tie -> smaller idx) ----
#pragma unroll
    for (int r = 0; r < 8; ++r) {
        float v = minv[r];
        int   ix = mini[r];
#pragma unroll
        for (int off = 8; off >= 1; off >>= 1) {
            float ov = __shfl_xor(v, off, 32);
            int   oi = __shfl_xor(ix, off, 32);
            if (ov < v || (ov == v && oi < ix)) { v = ov; ix = oi; }
        }
        if ((lane & 15) == 0) {
            int grow = rowBase + w * 16 + h * 8 + r;   // C layout: M = r (+8 hi half)
            idxo[grow] = ix;
        }
    }
}

// ---------------------------------------------------------------------------
// Kernel 3: gather z_q (exact fp32), loss partials, usage histogram, used flag.
// One wave per row; 8 rows per 256-thread block; grid = 4096.
// ---------------------------------------------------------------------------
__global__ __launch_bounds__(256) void vq_epilogue(const float* __restrict__ z,
                                                   const float* __restrict__ emb,
                                                   const int* __restrict__ idxo,
                                                   float* __restrict__ zq_out,
                                                   float* __restrict__ lossacc,
                                                   float* __restrict__ hist,
                                                   float* __restrict__ used) {
    const int lane = threadIdx.x & 31;
    const int w    = threadIdx.x >> 5;
    const int row  = blockIdx.x * 8 + w;
    const int idx  = idxo[row];

    const float4* zp = (const float4*)(z      + (size_t)row * E_DIM);
    const float4* ep = (const float4*)(emb    + (size_t)idx * E_DIM);
    float4*       op = (float4*)(zq_out + (size_t)row * E_DIM);

    float accl = 0.0f;
#pragma unroll
    for (int j = 0; j < 2; ++j) {
        float4 zv = zp[lane * 2 + j];
        float4 ev = ep[lane * 2 + j];
        op[lane * 2 + j] = ev;                 // z_q_st == z_q numerically
        float dx = ev.x - zv.x, dy = ev.y - zv.y, dz = ev.z - zv.z, dw = ev.w - zv.w;
        accl += dx*dx + dy*dy + dz*dz + dw*dw;
    }
#pragma unroll
    for (int off = 16; off >= 1; off >>= 1)
        accl += __shfl_xor(accl, off, 32);

    if (lane == 0) {
        atomicAdd(lossacc, accl);
        atomicAdd(&hist[idx], 1.0f);
        used[idx] = 1.0f;                      // racy same-value store: fine
    }
}

// ---------------------------------------------------------------------------
// Kernel 4: finalize age/usage/loss.
// ---------------------------------------------------------------------------
__global__ __launch_bounds__(256) void vq_stats(const float* __restrict__ code_age,
                                                const float* __restrict__ code_usage,
                                                const float* __restrict__ hist,
                                                const float* __restrict__ used,
                                                const float* __restrict__ lossacc,
                                                float* __restrict__ out) {
    const int n = blockIdx.x * 256 + threadIdx.x;
    out[AGE_OFF   + n] = (used[n] != 0.0f) ? 0.0f : code_age[n] + 1.0f;
    out[USAGE_OFF + n] = code_usage[n] + hist[n];
    if (n == 0)
        out[LOSS_OFF] = *lossacc * (1.25f / 8388608.0f);   // (1+beta) * sum / (M*D)
}

// ---------------------------------------------------------------------------
extern "C" void kernel_launch(void* const* d_in, const int* in_sizes, int n_in,
                              void* d_out, int out_size, void* d_ws, size_t ws_size,
                              hipStream_t stream) {
    (void)in_sizes; (void)n_in; (void)out_size; (void)ws_size;
    const float* z          = (const float*)d_in[0];
    const float* emb        = (const float*)d_in[1];
    const float* code_age   = (const float*)d_in[2];
    const float* code_usage = (const float*)d_in[3];
    float* out = (float*)d_out;
    char*  ws  = (char*)d_ws;

    unsigned short* embb    = (unsigned short*)(ws + WS_EMBB);
    float*          enorm   = (float*)(ws + WS_ENORM);
    float*          hist    = (float*)(ws + WS_HIST);
    float*          used    = (float*)(ws + WS_USED);
    float*          lossacc = (float*)(ws + WS_LOSS);
    int*            idxo    = (int*)(out + IDX_OFF);

    vq_prep    <<<N_E / 8,        256, 0, stream>>>(emb, embb, enorm, hist, used, lossacc);
    vq_argmin  <<<M_TOTAL / MB,   128, 0, stream>>>(z, embb, enorm, idxo);
    vq_epilogue<<<M_TOTAL / 8,    256, 0, stream>>>(z, emb, idxo, out, lossacc, hist, used);
    vq_stats   <<<N_E / 256,      256, 0, stream>>>(code_age, code_usage, hist, used, lossacc, out);
}
